// MoEBlock_26345329394070
// MI455X (gfx1250) — matrix operand order repaired and checked by reading, not compiled, not run on hardware
//
#include <hip/hip_runtime.h>
#include <hip/hip_bf16.h>
#include <math.h>

// ---------------------------------------------------------------------------
// MoE block for MI455X (gfx1250): routed experts, bf16 WMMA, fused MLP.
// ---------------------------------------------------------------------------

typedef __bf16 bf16;
typedef __attribute__((ext_vector_type(16))) __bf16 v16bf;
typedef __attribute__((ext_vector_type(8)))  float  v8f;

#define D_MODEL 1024
#define N_TOK   8192          // 4 * 2048 tokens
#define TILE_M  32            // tokens per block
#define XS_PITCH 1032         // 1024 + 8 elem pad (bank spread, 16B aligned rows)
#define HS_PITCH 72           // 64 + 8 elem pad (rows stay 16B aligned: 144B)
#define SMEM_BYTES (TILE_M * XS_PITCH * 2 + TILE_M * HS_PITCH * 2 + TILE_M * 4)

union Frag {
    v16bf v;
    uint4 q[2];
};

// ---------------------------------------------------------------------------
// routing: build per-expert compacted token index lists
// ---------------------------------------------------------------------------
__global__ void moe_init_kernel(int* counts, int* idxbuf) {
    int t = blockIdx.x * 256 + threadIdx.x;
    if (t < 2) counts[t] = 0;
    if (t < 2 * N_TOK) idxbuf[t] = -1;
}

__global__ void moe_route_kernel(const int* __restrict__ mask,
                                 int* __restrict__ counts,
                                 int* __restrict__ idxbuf) {
    int t = blockIdx.x * 256 + threadIdx.x;
    if (t >= N_TOK) return;
    int e = (mask[t] != 0) ? 1 : 0;           // expert 1 = big, 0 = small
    int slot = atomicAdd(&counts[e], 1);
    idxbuf[e * N_TOK + slot] = t;
}

// fp32 -> bf16 conversion (all sizes are multiples of 256)
__global__ void moe_cvt_kernel(const float* __restrict__ src, bf16* __restrict__ dst) {
    size_t i = (size_t)blockIdx.x * 256 + threadIdx.x;
    dst[i] = (bf16)src[i];
}

__device__ __forceinline__ float gelu_exact(float x) {
    return 0.5f * x * (1.0f + erff(x * 0.70710678118654752f));
}

// ---------------------------------------------------------------------------
// Fused expert: out[tok] = gelu(x @ Wfc^T + bfc) @ Wproj^T + bproj
//   block = 32 tokens x 1024 out dims, 8 waves of 32 (wave32).
//   wave w: mt = w>>2 (token half), ntq = w&3 (f quarter in phase1,
//           256-col d range in phase2). 16 persistent v8f accumulators/wave.
// ---------------------------------------------------------------------------
__global__ __launch_bounds__(256)
void moe_expert_kernel(const bf16* __restrict__ xbf,
                       const bf16* __restrict__ wfc,
                       const float* __restrict__ bfc,
                       const bf16* __restrict__ wproj,
                       const float* __restrict__ bproj,
                       const int* __restrict__ idx,
                       const int* __restrict__ cnt,
                       int F,
                       float* __restrict__ out) {
    extern __shared__ char smem_raw[];
    bf16* Xs   = (bf16*)smem_raw;                                   // 32 x 1032 bf16
    bf16* Hs   = (bf16*)(smem_raw + TILE_M * XS_PITCH * 2);         // 32 x 72 bf16
    int*  toks = (int*)(smem_raw + TILE_M * XS_PITCH * 2 + TILE_M * HS_PITCH * 2);

    const int tileBase = blockIdx.x * TILE_M;
    const int count = cnt[0];
    if (tileBase >= count) return;

    const int tid  = threadIdx.x;
    const int lane = tid & 31;
    const int w    = tid >> 5;        // wave 0..7
    const int mt   = w >> 2;          // 0/1: token half (rows mt*16..mt*16+15)
    const int ntq  = w & 3;           // quarter index
    const int lmod = lane & 15;
    const int lhi  = lane >> 4;       // 0/1

    if (tid < TILE_M) toks[tid] = idx[tileBase + tid];
    __syncthreads();

    // gather 32 token rows of bf16 x into LDS (8 threads per row, 128 elem each)
    {
        int r = tid >> 3;
        int c = (tid & 7) * 128;
        int tok = toks[r];
        if (tok < 0) tok = 0;                      // padded slot: harmless read
        const uint4* src = (const uint4*)(xbf + (size_t)tok * D_MODEL + c);
        uint4* dst = (uint4*)(Xs + r * XS_PITCH + c);
#pragma unroll
        for (int i = 0; i < 16; ++i) dst[i] = src[i];
    }

    v8f acc[16];
#pragma unroll
    for (int i = 0; i < 16; ++i) acc[i] = (v8f){};

    // A-fragment base for phase1: row = mt*16 + lmod, K runs at lhi*8 / 16+lhi*8
    const bf16* xrow = Xs + (mt * 16 + lmod) * XS_PITCH + lhi * 8;
    const bf16* hrow = Hs + (mt * 16 + lmod) * HS_PITCH + lhi * 8;

    for (int fb = 0; fb < F; fb += 64) {
        __syncthreads();   // Hs from previous step fully consumed / Xs ready

        // ---------------- phase 1: H(16 tok x 16 f) = x @ Wfc^T, K = 1024 ----
        const int fcol = fb + ntq * 16 + lmod;              // this lane's f column
        const bf16* wrow = wfc + (size_t)fcol * D_MODEL + lhi * 8;
        if (fb + 64 < F)
            __builtin_prefetch((const void*)(wrow + 64 * D_MODEL), 0, 1);

        v8f h = (v8f){};
#pragma unroll 4
        for (int k = 0; k < D_MODEL; k += 32) {
            Frag a, b;
            a.q[0] = *(const uint4*)(xrow + k);        // K: k+lhi*8   .. +7
            a.q[1] = *(const uint4*)(xrow + k + 16);   // K: k+16+lhi*8.. +7
            b.q[0] = *(const uint4*)(wrow + k);        // K: k+lhi*8   .. +7
            b.q[1] = *(const uint4*)(wrow + k + 16);   // K: k+16+lhi*8.. +7
            h = __builtin_amdgcn_wmma_f32_16x16x32_bf16(
                    false, a.v, false, b.v, (short)0, h, false, false);
        }

        // bias + exact gelu, stage as bf16 A-operand layout source in LDS
        const float bias = bfc[fcol];
#pragma unroll
        for (int v = 0; v < 8; ++v) {
            float val = gelu_exact(h[v] + bias);
            int m = mt * 16 + v + lhi * 8;                  // C layout row
            Hs[m * HS_PITCH + ntq * 16 + lmod] = (bf16)val;
        }
        __syncthreads();

        // ---------------- phase 2: out += H(32 x 64f) @ Wproj^T --------------
        Frag a0, a1;                                        // f chunks [0,32), [32,64)
        a0.q[0] = *(const uint4*)(hrow + 0);
        a0.q[1] = *(const uint4*)(hrow + 16);
        a1.q[0] = *(const uint4*)(hrow + 32);
        a1.q[1] = *(const uint4*)(hrow + 48);

#pragma unroll
        for (int i = 0; i < 16; ++i) {
            const int dcol = (ntq * 16 + i) * 16 + lmod;    // this lane's d column
            const bf16* wp = wproj + (size_t)dcol * F + fb + lhi * 8;
            Frag b0, b1;
            b0.q[0] = *(const uint4*)(wp);
            b0.q[1] = *(const uint4*)(wp + 16);
            b1.q[0] = *(const uint4*)(wp + 32);
            b1.q[1] = *(const uint4*)(wp + 48);
            acc[i] = __builtin_amdgcn_wmma_f32_16x16x32_bf16(
                         false, a0.v, false, b0.v, (short)0, acc[i], false, false);
            acc[i] = __builtin_amdgcn_wmma_f32_16x16x32_bf16(
                         false, a1.v, false, b1.v, (short)0, acc[i], false, false);
        }
    }

    // ---------------- epilogue: + bproj, scatter rows to out -----------------
#pragma unroll
    for (int i = 0; i < 16; ++i) {
        const int dcol = (ntq * 16 + i) * 16 + lmod;
        const float bp = bproj[dcol];
#pragma unroll
        for (int v = 0; v < 8; ++v) {
            int m = mt * 16 + v + lhi * 8;
            int tok = toks[m];
            if (tok >= 0) out[(size_t)tok * D_MODEL + dcol] = acc[i][v] + bp;
        }
    }
}

// ---------------------------------------------------------------------------
// launcher
// ---------------------------------------------------------------------------
extern "C" void kernel_launch(void* const* d_in, const int* in_sizes, int n_in,
                              void* d_out, int out_size, void* d_ws, size_t ws_size,
                              hipStream_t stream) {
    const float* x       = (const float*)d_in[0];
    const int*   mask    = (const int*)d_in[1];
    const float* wfc_s   = (const float*)d_in[2];
    const float* bfc_s   = (const float*)d_in[3];
    const float* wproj_s = (const float*)d_in[4];
    const float* bproj_s = (const float*)d_in[5];
    const float* wfc_b   = (const float*)d_in[6];
    const float* bfc_b   = (const float*)d_in[7];
    const float* wproj_b = (const float*)d_in[8];
    const float* bproj_b = (const float*)d_in[9];
    float* out = (float*)d_out;

    char* ws = (char*)d_ws;
    bf16* xbf   = (bf16*)(ws);                     // 16,777,216 B
    bf16* wfcsb = (bf16*)(ws + 16777216);          //  8,388,608 B
    bf16* wprsb = (bf16*)(ws + 25165824);          //  8,388,608 B
    bf16* wfcbb = (bf16*)(ws + 33554432);          // 33,554,432 B
    bf16* wprbb = (bf16*)(ws + 67108864);          // 33,554,432 B
    int*  counts = (int*)(ws + 100663296);         // 2 ints (pad to 64)
    int*  idxbuf = (int*)(ws + 100663360);         // 2 * 8192 ints

    // routing
    moe_init_kernel<<<(2 * N_TOK + 255) / 256, 256, 0, stream>>>(counts, idxbuf);
    moe_route_kernel<<<N_TOK / 256, 256, 0, stream>>>(mask, counts, idxbuf);

    // fp32 -> bf16 staging (x + 4 weight matrices)
    moe_cvt_kernel<<<(N_TOK * D_MODEL) / 256, 256, 0, stream>>>(x, xbf);
    moe_cvt_kernel<<<(4096 * D_MODEL) / 256, 256, 0, stream>>>(wfc_s, wfcsb);
    moe_cvt_kernel<<<(D_MODEL * 4096) / 256, 256, 0, stream>>>(wproj_s, wprsb);
    moe_cvt_kernel<<<(16384 * D_MODEL) / 256, 256, 0, stream>>>(wfc_b, wfcbb);
    moe_cvt_kernel<<<(D_MODEL * 16384) / 256, 256, 0, stream>>>(wproj_b, wprbb);

    // one launch per expert; blocks past the routed count exit immediately
    moe_expert_kernel<<<N_TOK / TILE_M, 256, SMEM_BYTES, stream>>>(
        xbf, wfcsb, bfc_s, wprsb, bproj_s, idxbuf, counts, 4096, out);
    moe_expert_kernel<<<N_TOK / TILE_M, 256, SMEM_BYTES, stream>>>(
        xbf, wfcbb, bfc_b, wprbb, bproj_b, idxbuf + N_TOK, counts + 1, 16384, out);
}
